// SparseVectorQuantizer_3504693314202
// MI455X (gfx1250) — matrix operand order repaired and checked by reading, not compiled, not run on hardware
//
#include <hip/hip_runtime.h>
#include <hip/hip_bf16.h>

// ---------------------------------------------------------------------------
// VQ encode on MI455X (gfx1250), split-bf16 compensated GEMM:
//   x = hi + lo (both bf16):  S = Zhi*Ehi^T + Zlo*Ehi^T + Zhi*Elo^T  (f32 acc)
//   -> f32-level accuracy (err ~3e-5) at 6 bf16 WMMAs per 16x16 tile.
//   argmin_k ( e_sq[k] - 2*S[n,k] )   (z_sq is row-constant -> drops out)
// Codebook chunks are double-buffered in LDS and staged with
// GLOBAL_LOAD_ASYNC_TO_LDS_B128 (ASYNCcnt) so the copy overlaps the WMMAs.
// Outputs (flat, reference return order):
//   [0, 2097152)          quantized_st (== embeddings[idx], exact f32 gather)
//   [2097152]             vq_loss, [2097153] commitment (same forward value)
//   [2097154, +32768)     indices as float
// ---------------------------------------------------------------------------

#define N_VOXELS  32768
#define N_CODES   8192
#define EMBED_DIM 64
#define OUT_Q     (N_VOXELS * EMBED_DIM)   // 2097152

typedef __attribute__((ext_vector_type(16))) __bf16 v16bf;
typedef __attribute__((ext_vector_type(8)))  float  v8f;

#define CHUNK   128           // codes staged in LDS per buffer
#define NCHUNK  (N_CODES / CHUNK)   // 64
#define TILES   (CHUNK / 16)        // 8
#define LDSROW  72            // bf16 elems per LDS row (64 + 8 pad -> 144B,
                              // 36-bank stride => conflict-free b128 reads)

#if __has_builtin(__builtin_amdgcn_global_load_async_to_lds_b128)
#define HAVE_ASYNC_LDS 1
typedef int v4i __attribute__((vector_size(16)));
typedef __attribute__((address_space(1))) v4i* gv4i_p;   // global src
typedef __attribute__((address_space(3))) v4i* lv4i_p;   // LDS dst
#define ASYNC_CP16(dstLds, srcG)                                            \
    __builtin_amdgcn_global_load_async_to_lds_b128(                         \
        (gv4i_p)(void*)(srcG), (lv4i_p)(dstLds), 0, 0)
#if __has_builtin(__builtin_amdgcn_s_wait_asynccnt)
#define WAIT_ASYNC() __builtin_amdgcn_s_wait_asynccnt(0)
#else
#define WAIT_ASYNC() asm volatile("s_wait_asynccnt 0x0" ::: "memory")
#endif
#else
#define HAVE_ASYNC_LDS 0
#define WAIT_ASYNC() ((void)0)
#endif

// ---------------------------------------------------------------------------
// Kernel 1: E (f32) -> E_hi, E_lo (bf16) workspace, plus exact f32 e_sq[k].
// 8 waves per block, one code row per wave. grid = 8192/8 = 1024.
// ---------------------------------------------------------------------------
__global__ __launch_bounds__(256) void vq_prep_kernel(
    const float* __restrict__ emb, __bf16* __restrict__ ebf_h,
    __bf16* __restrict__ ebf_l, float* __restrict__ esq)
{
    int k    = blockIdx.x * 8 + (threadIdx.x >> 5);
    int lane = threadIdx.x & 31;
    const float* er = emb + k * EMBED_DIM;
    float s = 0.0f;
#pragma unroll
    for (int q = 0; q < 2; ++q) {
        float v = er[lane + q * 32];
        __bf16 hi = (__bf16)v;
        __bf16 lo = (__bf16)(v - (float)hi);
        ebf_h[k * EMBED_DIM + lane + q * 32] = hi;
        ebf_l[k * EMBED_DIM + lane + q * 32] = lo;
        s = fmaf(v, v, s);
    }
#pragma unroll
    for (int off = 16; off >= 1; off >>= 1)
        s += __shfl_xor(s, off, 32);
    if (lane == 0) esq[k] = s;
}

// ---------------------------------------------------------------------------
// Kernel 2: split-bf16 WMMA distance GEMM + running argmin.
// Block = 256 threads = 8 waves; each wave owns 16 rows (A hi/lo in regs).
// 64 chunks of 128 codes, double-buffered async staging. grid = 256.
// ---------------------------------------------------------------------------
__global__ __launch_bounds__(256) void vq_argmin_kernel(
    const float*  __restrict__ z,
    const __bf16* __restrict__ ebf_h,
    const __bf16* __restrict__ ebf_l,
    const float*  __restrict__ esq,
    int* __restrict__ bestIdxOut)
{
    __shared__ __bf16 ldsEh[2][CHUNK * LDSROW];   // 2 x 18 KB
    __shared__ __bf16 ldsEl[2][CHUNK * LDSROW];   // 2 x 18 KB
    __shared__ float  ldsQ[2][CHUNK];             // 1 KB

    const int tid  = threadIdx.x;
    const int lane = tid & 31;
    const int wave = tid >> 5;
    const int hw   = lane >> 4;       // half-wave (K sub-block: K = 16*hw..16*hw+15, contiguous)
    const int m    = lane & 15;       // row-in-tile (A) / col-in-tile (B)
    const int rowBase = blockIdx.x * 128 + wave * 16;

    // stage one chunk (hi + lo rows + e_sq) into LDS buffer `buf`
    auto stage = [&](int buf, int cBase) {
        const int r    = tid >> 1;        // 0..127: code row within chunk
        const int hsel = tid & 1;         // which 64B half of the row
        const __bf16* gh = ebf_h + (cBase + r) * EMBED_DIM + hsel * 32;
        const __bf16* gl = ebf_l + (cBase + r) * EMBED_DIM + hsel * 32;
        __bf16* dh = &ldsEh[buf][r * LDSROW + hsel * 32];
        __bf16* dl = &ldsEl[buf][r * LDSROW + hsel * 32];
#if HAVE_ASYNC_LDS
#pragma unroll
        for (int q = 0; q < 4; ++q) {     // 4 x 16B per array
            ASYNC_CP16(dh + q * 8, gh + q * 8);
            ASYNC_CP16(dl + q * 8, gl + q * 8);
        }
#else
#pragma unroll
        for (int q = 0; q < 4; ++q) {
            ((float4*)dh)[q] = ((const float4*)gh)[q];
            ((float4*)dl)[q] = ((const float4*)gl)[q];
        }
#endif
        if (tid < CHUNK) ldsQ[buf][tid] = esq[cBase + tid];
    };

    // ---- A fragments (16x32 bf16 each), contiguous K layout (matches B):
    // lanes 0-15 : elems 0..15 -> K=0..15
    // lanes 16-31: elems 0..15 -> K=16..31
    stage(0, 0);   // kick off first chunk before the (independent) A loads

    v16bf A0h, A1h, A0l, A1l;
    {
        const float* zr = z + (rowBase + m) * EMBED_DIM;
#pragma unroll
        for (int e = 0; e < 8; ++e) {
            float v0 = zr[hw * 16 + e];
            float v1 = zr[hw * 16 + 8 + e];
            float v2 = zr[32 + hw * 16 + e];
            float v3 = zr[32 + hw * 16 + 8 + e];
            __bf16 h0 = (__bf16)v0, h1 = (__bf16)v1;
            __bf16 h2 = (__bf16)v2, h3 = (__bf16)v3;
            A0h[e] = h0;     A0l[e]     = (__bf16)(v0 - (float)h0);
            A0h[e + 8] = h1; A0l[e + 8] = (__bf16)(v1 - (float)h1);
            A1h[e] = h2;     A1l[e]     = (__bf16)(v2 - (float)h2);
            A1h[e + 8] = h3; A1l[e + 8] = (__bf16)(v3 - (float)h3);
        }
    }

    float bestV[8];
    int   bestI[8];
#pragma unroll
    for (int j = 0; j < 8; ++j) { bestV[j] = 3.4e38f; bestI[j] = 0; }

    WAIT_ASYNC();
    __syncthreads();

    for (int chunk = 0; chunk < NCHUNK; ++chunk) {
        const int buf   = chunk & 1;
        const int cBase = chunk * CHUNK;
        if (chunk + 1 < NCHUNK)
            stage(buf ^ 1, cBase + CHUNK);   // async copy overlaps the WMMAs

        // ---- 8 code-tiles of 16 codes each, 6 WMMAs per tile
#pragma unroll 2
        for (int t = 0; t < TILES; ++t) {
            const int cl = t * 16 + m;                 // this lane's code column
            // B (32x16 bf16): lane = column, elem e -> K = hw*16 + e
            const __bf16* bh = &ldsEh[buf][cl * LDSROW + hw * 16];
            const __bf16* bl = &ldsEl[buf][cl * LDSROW + hw * 16];
            v16bf B0h = *(const v16bf*)(bh);
            v16bf B1h = *(const v16bf*)(bh + 32);
            v16bf B0l = *(const v16bf*)(bl);
            v16bf B1l = *(const v16bf*)(bl + 32);

            v8f acc = {};
            acc = __builtin_amdgcn_wmma_f32_16x16x32_bf16(
                      false, A0h, false, B0h, (short)0, acc, false, false);
            acc = __builtin_amdgcn_wmma_f32_16x16x32_bf16(
                      false, A1h, false, B1h, (short)0, acc, false, false);
            acc = __builtin_amdgcn_wmma_f32_16x16x32_bf16(
                      false, A0l, false, B0h, (short)0, acc, false, false);
            acc = __builtin_amdgcn_wmma_f32_16x16x32_bf16(
                      false, A1l, false, B1h, (short)0, acc, false, false);
            acc = __builtin_amdgcn_wmma_f32_16x16x32_bf16(
                      false, A0h, false, B0l, (short)0, acc, false, false);
            acc = __builtin_amdgcn_wmma_f32_16x16x32_bf16(
                      false, A1h, false, B1l, (short)0, acc, false, false);

            const float eq = ldsQ[buf][cl];
            const int   c  = cBase + cl;
#pragma unroll
            for (int j = 0; j < 8; ++j) {
                float score = fmaf(-2.0f, acc[j], eq);    // d^2 - ||z||^2
                if (score < bestV[j]) { bestV[j] = score; bestI[j] = c; }
            }
        }

        WAIT_ASYNC();      // this wave's copies for chunk+1 have landed
        __syncthreads();   // publish; all waves done reading buf
    }

    // ---- argmin reduce across the 16 lanes of each half-wave.
    // C/D layout: VGPR j, lanes 0-15 -> row j; lanes 16-31 -> row j+8.
#pragma unroll
    for (int j = 0; j < 8; ++j) {
        float v  = bestV[j];
        int   bi = bestI[j];
#pragma unroll
        for (int off = 8; off >= 1; off >>= 1) {          // stays inside 16-lane groups
            float ov = __shfl_xor(v, off, 32);
            int   oi = __shfl_xor(bi, off, 32);
            if (ov < v || (ov == v && oi < bi)) { v = ov; bi = oi; }
        }
        if (m == 0)
            bestIdxOut[rowBase + hw * 8 + j] = bi;
    }
}

// ---------------------------------------------------------------------------
// Kernel 3: gather f32 code rows -> quantized_st, float indices,
// deterministic per-block partial sums of (z - q)^2. grid = 2048.
// ---------------------------------------------------------------------------
__global__ __launch_bounds__(256) void vq_gather_kernel(
    const float* __restrict__ z, const float* __restrict__ emb,
    const int* __restrict__ bestIdx, float* __restrict__ out,
    float* __restrict__ partials)
{
    __shared__ float red[256];
    const int tid  = threadIdx.x;
    const int rloc = tid >> 4;
    const int c4   = (tid & 15) * 4;
    const int row  = blockIdx.x * 16 + rloc;
    const int idx  = bestIdx[row];

    const float4 zv = *(const float4*)(z   + row * EMBED_DIM + c4);
    const float4 ev = *(const float4*)(emb + idx * EMBED_DIM + c4);
    *(float4*)(out + row * EMBED_DIM + c4) = ev;   // straight-through fwd == quantized

    float dx = zv.x - ev.x, dy = zv.y - ev.y, dz = zv.z - ev.z, dw = zv.w - ev.w;
    red[tid] = dx * dx + dy * dy + dz * dz + dw * dw;
    if ((tid & 15) == 0) out[OUT_Q + 2 + row] = (float)idx;
    __syncthreads();
#pragma unroll
    for (int s = 128; s > 0; s >>= 1) {
        if (tid < s) red[tid] += red[tid + s];
        __syncthreads();
    }
    if (tid == 0) partials[blockIdx.x] = red[0];
}

// ---------------------------------------------------------------------------
// Kernel 4: deterministic final reduction -> both loss scalars.
// ---------------------------------------------------------------------------
__global__ __launch_bounds__(256) void vq_finalize_kernel(
    const float* __restrict__ partials, float* __restrict__ out)
{
    __shared__ float red[256];
    float s = 0.0f;
    for (int i = threadIdx.x; i < 2048; i += 256) s += partials[i];
    red[threadIdx.x] = s;
    __syncthreads();
#pragma unroll
    for (int st = 128; st > 0; st >>= 1) {
        if (threadIdx.x < st) red[threadIdx.x] += red[threadIdx.x + st];
        __syncthreads();
    }
    if (threadIdx.x == 0) {
        float mse = red[0] / (float)(N_VOXELS * EMBED_DIM);
        out[OUT_Q]     = mse;   // vq_loss
        out[OUT_Q + 1] = mse;   // commitment_loss (same forward value)
    }
}

// ---------------------------------------------------------------------------
extern "C" void kernel_launch(void* const* d_in, const int* in_sizes, int n_in,
                              void* d_out, int out_size, void* d_ws, size_t ws_size,
                              hipStream_t stream) {
    const float* z   = (const float*)d_in[0];   // [32768, 64]
    const float* emb = (const float*)d_in[1];   // [8192, 64]
    float* out = (float*)d_out;

    char* ws = (char*)d_ws;
    __bf16* ebf_h    = (__bf16*)(ws);                           // 1,048,576 B
    __bf16* ebf_l    = (__bf16*)(ws + 1048576);                 // 1,048,576 B
    float*  esq      = (float*)(ws + 2097152);                  //    32,768 B
    int*    bestIdx  = (int*)  (ws + 2097152 + 32768);          //   131,072 B
    float*  partials = (float*)(ws + 2097152 + 32768 + 131072); //     8,192 B

    vq_prep_kernel    <<<N_CODES / 8,    256, 0, stream>>>(emb, ebf_h, ebf_l, esq);
    vq_argmin_kernel  <<<N_VOXELS / 128, 256, 0, stream>>>(z, ebf_h, ebf_l, esq, bestIdx);
    vq_gather_kernel  <<<N_VOXELS / 16,  256, 0, stream>>>(z, emb, bestIdx, out, partials);
    vq_finalize_kernel<<<1,              256, 0, stream>>>(partials, out);
}
